// GraphNet_73632919322703
// MI455X (gfx1250) — compile-verified, matrix operand order checked by reading, not run on hardware
//
#include <hip/hip_runtime.h>

typedef unsigned short u16t;
typedef __attribute__((ext_vector_type(16))) __bf16 bf16x16;
typedef __attribute__((ext_vector_type(8)))  float  f32x8;

#define O_N   25000
#define T_N   100000
#define H_N   512
#define DOUT  128
#define M_N   1152
#define KX    448      // padded concat K: 160 + 128 + 160
#define OBJP  160      // padded node feature width (143 or 128 real)
#define BT    32       // edges per block in the fused gconv kernel

union Frag { bf16x16 v; uint4 q[2]; };

__device__ __forceinline__ u16t f2bf(float f) {
  unsigned int x = __float_as_uint(f);
  unsigned int r = x + 0x7FFFu + ((x >> 16) & 1u);   // round-to-nearest-even
  return (u16t)(r >> 16);
}

__device__ __forceinline__ f32x8 fzero() {
  f32x8 z;
#pragma unroll
  for (int i = 0; i < 8; ++i) z[i] = 0.0f;
  return z;
}

__device__ __forceinline__ f32x8 wmma_bf16(Frag a, Frag b, f32x8 c) {
  // D(16x16,f32) = A(16x32,bf16) x B(32x16,bf16) + C
  return __builtin_amdgcn_wmma_f32_16x16x32_bf16(false, a.v, false, b.v,
                                                 (short)0, c, false, false);
}

__device__ __forceinline__ Frag load_frag(const u16t* p) {
  // two contiguous 8xbf16 (16B) chunks: K{0..7}+8*half and K{16..23}+8*half
  Frag f;
  f.q[0] = *(const uint4*)(p);
  f.q[1] = *(const uint4*)(p + 16);
  return f;
}

// Native CDNA5 no-return FP32 global atomic add (STOREcnt-tracked; S_ENDPGM's
// implicit wait-idle guarantees completion before kernel end).
__device__ __forceinline__ void atomic_add_f32(float* p, float v) {
  asm volatile("global_atomic_add_f32 %0, %1, off" : : "v"(p), "v"(v) : "memory");
}

// ---------------------------------------------------------------- prep kernels

// WT[n*K + k] = bf16(W[k*N + n])
__global__ void k_transpose(const float* __restrict__ W, u16t* __restrict__ WT,
                            int K, int N) {
  int idx = blockIdx.x * 256 + threadIdx.x;
  if (idx >= K * N) return;
  int n = idx / K, k = idx - n * K;
  WT[idx] = f2bf(W[k * N + n]);
}

// Build padded/transposed W1: source rows [0,lenObj)|[lenObj,lenObj+128)|[..+lenObj)
// mapped to K segments [0,160)|[160,288)|[288,448), zero-padded, N-major output.
__global__ void k_build_w1t(const float* __restrict__ W1, u16t* __restrict__ WT,
                            int lenObj) {
  int idx = blockIdx.x * 256 + threadIdx.x;
  if (idx >= 512 * KX) return;
  int n = idx / KX, k = idx - n * KX;
  float v = 0.0f;
  if (k < 160)      { if (k < lenObj) v = W1[k * 512 + n]; }
  else if (k < 288) { v = W1[(lenObj + (k - 160)) * 512 + n]; }
  else              { int kk = k - 288; if (kk < lenObj) v = W1[(lenObj + 128 + kk) * 512 + n]; }
  WT[idx] = f2bf(v);
}

__global__ void k_zero_f32(float* __restrict__ p, int n) {
  int i = blockIdx.x * 256 + threadIdx.x;
  if (i < n) p[i] = 0.0f;
}

__global__ void k_counts(const int* __restrict__ edges, float* __restrict__ cnt) {
  int t = blockIdx.x * 256 + threadIdx.x;
  if (t < T_N) {
    atomic_add_f32(&cnt[edges[2 * t + 0]], 1.0f);
    atomic_add_f32(&cnt[edges[2 * t + 1]], 1.0f);
  }
}

__global__ void k_invert(float* __restrict__ c) {
  int i = blockIdx.x * 256 + threadIdx.x;
  if (i < O_N) c[i] = 1.0f / fmaxf(c[i], 1.0f);
}

__global__ void k_nodefeat0(const float* __restrict__ objs, const float* __restrict__ pc,
                            u16t* __restrict__ objf) {
  int idx = blockIdx.x * 256 + threadIdx.x;
  if (idx >= O_N * OBJP) return;
  int i = idx / OBJP, j = idx - i * OBJP;
  float v = 0.0f;
  if (j < 128)      v = pc[i * 128 + j];
  else if (j < 143) v = objs[i * 15 + (j - 128)];
  objf[idx] = f2bf(v);
}

__global__ void k_edgefeat0(const int* __restrict__ state, const int* __restrict__ direct,
                            const float* __restrict__ es, const float* __restrict__ ed,
                            u16t* __restrict__ ef) {
  int idx = blockIdx.x * 256 + threadIdx.x;
  if (idx >= T_N * 128) return;
  int t = idx >> 7, j = idx & 127;
  float v = (j < 64) ? es[state[t] * 64 + j] : ed[direct[t] * 64 + (j - 64)];
  ef[idx] = f2bf(v);
}

// ------------------------------------------------------- fused edge gconv (WMMA)
// 32 edges per block: each B-fragment load feeds TWO wmma ops (two M-subtiles),
// halving L2 weight traffic vs a 16-row tile.

__global__ __launch_bounds__(256) void k_edge_gconv(
    const u16t* __restrict__ objf,      // O x 160 bf16 (padded)
    const u16t* __restrict__ edgef,     // T x 128 bf16
    const int*  __restrict__ edges,     // T x 2
    const u16t* __restrict__ W1T,       // 512 x 448 bf16 (N-major, padded K)
    const float* __restrict__ b1,       // 512
    const u16t* __restrict__ W2T,       // 1152 x 512 bf16 (N-major)
    const float* __restrict__ b2,       // 1152
    float* __restrict__ pooled,         // O x 512 f32 (atomic accum)
    u16t* __restrict__ edgef_out)       // T x 128 bf16
{
  __shared__ __align__(16) u16t sX[BT * KX];    // 28.7 KB
  __shared__ __align__(16) u16t sH[BT * H_N];   // 32.8 KB
  __shared__ int sS[BT], sO[BT];

  const int t0  = blockIdx.x * BT;
  const int tid = threadIdx.x;

  if (tid < BT) {
    sS[tid] = edges[(t0 + tid) * 2 + 0];
    sO[tid] = edges[(t0 + tid) * 2 + 1];
  }
  __syncthreads();

  // gather A-tile: BT rows x 56 chunks of 8 bf16
  for (int e = tid; e < BT * 56; e += 256) {
    int r = e / 56, c8 = e - r * 56;
    uint4 val;
    if (c8 < 20)      val = *(const uint4*)(objf + (size_t)sS[r] * OBJP + c8 * 8);
    else if (c8 < 36) val = *(const uint4*)(edgef + (size_t)(t0 + r) * 128 + (c8 - 20) * 8);
    else              val = *(const uint4*)(objf + (size_t)sO[r] * OBJP + (c8 - 36) * 8);
    *(uint4*)(sX + r * KX + c8 * 8) = val;
  }
  __syncthreads();

  const int wv = tid >> 5;
  const int lane = tid & 31;
  const int hl = lane >> 4;
  const int lr = lane & 15;

  // ---- GEMM1: H(32x512) = relu(X(32x448) @ W1 + b1)
  {
    f32x8 acc[2][4];
#pragma unroll
    for (int mt = 0; mt < 2; ++mt)
#pragma unroll
      for (int j = 0; j < 4; ++j) acc[mt][j] = fzero();

    for (int k0 = 0; k0 < KX; k0 += 32) {
      Frag a0 = load_frag(sX + (size_t)lr * KX + k0 + hl * 8);
      Frag a1 = load_frag(sX + (size_t)(16 + lr) * KX + k0 + hl * 8);
#pragma unroll
      for (int j = 0; j < 4; ++j) {
        const int n = wv * 64 + j * 16 + lr;
        Frag b = load_frag(W1T + (size_t)n * KX + k0 + hl * 8);
        acc[0][j] = wmma_bf16(a0, b, acc[0][j]);
        acc[1][j] = wmma_bf16(a1, b, acc[1][j]);
      }
    }
#pragma unroll
    for (int j = 0; j < 4; ++j) {
      const int n = wv * 64 + j * 16 + lr;
      const float bias = b1[n];
#pragma unroll
      for (int mt = 0; mt < 2; ++mt)
#pragma unroll
        for (int v = 0; v < 8; ++v) {
          float x = acc[mt][j][v] + bias;
          x = x > 0.0f ? x : 0.0f;
          sH[(size_t)(mt * 16 + v + hl * 8) * H_N + n] = f2bf(x);
        }
    }
  }
  __syncthreads();

  // ---- GEMM2: t(32x1152) = relu(H @ W2 + b2); split + scatter
  {
    f32x8 acc[2][9];
#pragma unroll
    for (int mt = 0; mt < 2; ++mt)
#pragma unroll
      for (int j = 0; j < 9; ++j) acc[mt][j] = fzero();

    for (int k0 = 0; k0 < H_N; k0 += 32) {
      Frag a0 = load_frag(sH + (size_t)lr * H_N + k0 + hl * 8);
      Frag a1 = load_frag(sH + (size_t)(16 + lr) * H_N + k0 + hl * 8);
#pragma unroll
      for (int j = 0; j < 9; ++j) {
        const int n = (wv * 9 + j) * 16 + lr;
        Frag b = load_frag(W2T + (size_t)n * H_N + k0 + hl * 8);
        acc[0][j] = wmma_bf16(a0, b, acc[0][j]);
        acc[1][j] = wmma_bf16(a1, b, acc[1][j]);
      }
    }
#pragma unroll
    for (int j = 0; j < 9; ++j) {
      const int n = (wv * 9 + j) * 16 + lr;
      const float bias = b2[n];
#pragma unroll
      for (int mt = 0; mt < 2; ++mt)
#pragma unroll
        for (int v = 0; v < 8; ++v) {
          float x = acc[mt][j][v] + bias;
          x = x > 0.0f ? x : 0.0f;
          const int r = mt * 16 + v + hl * 8;
          if (n < 512) {
            atomic_add_f32(pooled + (size_t)sS[r] * H_N + n, x);
          } else if (n < 640) {
            edgef_out[(size_t)(t0 + r) * 128 + (n - 512)] = f2bf(x);
          } else {
            atomic_add_f32(pooled + (size_t)sO[r] * H_N + (n - 640), x);
          }
        }
    }
  }
}

// ------------------------------------------------------------ node MLP (WMMA)

__global__ __launch_bounds__(256) void k_node_mlp(
    const float* __restrict__ pooled,   // O x 512
    const float* __restrict__ invc,     // O
    const u16t* __restrict__ V1T,       // 512 x 512 bf16 N-major
    const float* __restrict__ c1,
    const u16t* __restrict__ V2T,       // 128 x 512 bf16 N-major
    const float* __restrict__ c2,
    u16t* __restrict__ objf_out)        // O x 160 bf16 padded
{
  __shared__ __align__(16) u16t sX[16 * H_N];
  __shared__ __align__(16) u16t sH[16 * H_N];

  const int i0  = blockIdx.x * 16;
  const int tid = threadIdx.x;

  for (int e = tid; e < 16 * H_N; e += 256) {
    int r = e >> 9, c = e & 511;
    int i = i0 + r;
    float v = 0.0f;
    if (i < O_N) v = pooled[(size_t)i * H_N + c] * invc[i];
    sX[r * H_N + c] = f2bf(v);
  }
  __syncthreads();

  const int wv = tid >> 5;
  const int lane = tid & 31;
  const int hl = lane >> 4;
  const int lr = lane & 15;

  // GEMM1: relu(P @ V1 + c1) -> sH
  {
    f32x8 acc[4];
#pragma unroll
    for (int j = 0; j < 4; ++j) acc[j] = fzero();
    for (int k0 = 0; k0 < H_N; k0 += 32) {
      Frag a = load_frag(sX + (size_t)lr * H_N + k0 + hl * 8);
#pragma unroll
      for (int j = 0; j < 4; ++j) {
        const int n = wv * 64 + j * 16 + lr;
        Frag b = load_frag(V1T + (size_t)n * H_N + k0 + hl * 8);
        acc[j] = wmma_bf16(a, b, acc[j]);
      }
    }
#pragma unroll
    for (int j = 0; j < 4; ++j) {
      const int n = wv * 64 + j * 16 + lr;
      const float bias = c1[n];
#pragma unroll
      for (int v = 0; v < 8; ++v) {
        float x = acc[j][v] + bias;
        x = x > 0.0f ? x : 0.0f;
        sH[(size_t)(v + hl * 8) * H_N + n] = f2bf(x);
      }
    }
  }
  __syncthreads();

  // GEMM2: relu(H @ V2 + c2) -> objf_out (1 N-tile per wave)
  {
    f32x8 acc = fzero();
    for (int k0 = 0; k0 < H_N; k0 += 32) {
      Frag a = load_frag(sH + (size_t)lr * H_N + k0 + hl * 8);
      Frag b = load_frag(V2T + (size_t)(wv * 16 + lr) * H_N + k0 + hl * 8);
      acc = wmma_bf16(a, b, acc);
    }
    const int n = wv * 16 + lr;
    const float bias = c2[n];
#pragma unroll
    for (int v = 0; v < 8; ++v) {
      float x = acc[v] + bias;
      x = x > 0.0f ? x : 0.0f;
      const int i = i0 + v + hl * 8;
      if (i < O_N) objf_out[(size_t)i * OBJP + n] = f2bf(x);
    }
  }

  // zero padded tail cols [128,160)
  for (int e = tid; e < 16 * 32; e += 256) {
    int r = e >> 5, c = 128 + (e & 31);
    int i = i0 + r;
    if (i < O_N) objf_out[(size_t)i * OBJP + c] = 0;
  }
}

// ------------------------------------------------------- final edge head (WMMA)

__device__ __forceinline__ void head_stage(const u16t* sIn, const u16t* __restrict__ WT,
                                           const float* __restrict__ bias, u16t* sOut,
                                           int wv, int hl, int lr) {
  f32x8 acc = fzero();
  for (int k0 = 0; k0 < 128; k0 += 32) {
    Frag a = load_frag(sIn + (size_t)lr * 128 + k0 + hl * 8);
    Frag b = load_frag(WT + (size_t)(wv * 16 + lr) * 128 + k0 + hl * 8);
    acc = wmma_bf16(a, b, acc);
  }
  const int n = wv * 16 + lr;
  const float bb = bias[n];
#pragma unroll
  for (int v = 0; v < 8; ++v) {
    float x = acc[v] + bb;
    x = x > 0.0f ? x : 0.01f * x;            // leaky_relu
    sOut[(size_t)(v + hl * 8) * 128 + n] = f2bf(x);
  }
}

__global__ __launch_bounds__(256) void k_edge_head(
    const u16t* __restrict__ edgef,     // T x 128 bf16
    const u16t* __restrict__ L1T, const float* __restrict__ lb1,
    const u16t* __restrict__ L2T, const float* __restrict__ lb2,
    const u16t* __restrict__ L3T, const float* __restrict__ lb3,
    float* __restrict__ out)            // T x 16 f32
{
  __shared__ __align__(16) u16t sX[16 * 128];
  __shared__ __align__(16) u16t sH1[16 * 128];
  __shared__ __align__(16) u16t sH2[16 * 128];

  const int t0  = blockIdx.x * 16;
  const int tid = threadIdx.x;
  {
    int r = tid >> 4, c8 = tid & 15;   // exactly 256 16B chunks
    *(uint4*)(sX + r * 128 + c8 * 8) =
        *(const uint4*)(edgef + (size_t)(t0 + r) * 128 + c8 * 8);
  }
  __syncthreads();

  const int wv = tid >> 5;
  const int lane = tid & 31;
  const int hl = lane >> 4;
  const int lr = lane & 15;

  head_stage(sX, L1T, lb1, sH1, wv, hl, lr);
  __syncthreads();
  head_stage(sH1, L2T, lb2, sH2, wv, hl, lr);
  __syncthreads();

  if (wv == 0) {  // wave-uniform: EXEC all ones inside
    f32x8 acc = fzero();
    for (int k0 = 0; k0 < 128; k0 += 32) {
      Frag a = load_frag(sH2 + (size_t)lr * 128 + k0 + hl * 8);
      Frag b = load_frag(L3T + (size_t)lr * 128 + k0 + hl * 8);
      acc = wmma_bf16(a, b, acc);
    }
    const int n = lr;       // N = 16
    const float bb = lb3[n];
#pragma unroll
    for (int v = 0; v < 8; ++v)
      out[(size_t)(t0 + v + hl * 8) * 16 + n] = acc[v] + bb;
  }
}

// ----------------------------------------------------------------- host launch

extern "C" void kernel_launch(void* const* d_in, const int* in_sizes, int n_in,
                              void* d_out, int out_size, void* d_ws, size_t ws_size,
                              hipStream_t stream) {
  (void)in_sizes; (void)n_in; (void)out_size; (void)ws_size;
  const float* objs      = (const float*)d_in[0];
  const float* pc        = (const float*)d_in[1];
  const int*   edges     = (const int*)  d_in[2];
  const int*   state     = (const int*)  d_in[3];
  const int*   direct    = (const int*)  d_in[4];
  const float* emb_state = (const float*)d_in[5];
  const float* emb_drt   = (const float*)d_in[6];
  const float* g0_w1 = (const float*)d_in[7];
  const float* g0_b1 = (const float*)d_in[8];
  const float* g0_w2 = (const float*)d_in[9];
  const float* g0_b2 = (const float*)d_in[10];
  const float* g0_v1 = (const float*)d_in[11];
  const float* g0_c1 = (const float*)d_in[12];
  const float* g0_v2 = (const float*)d_in[13];
  const float* g0_c2 = (const float*)d_in[14];
  const float* gl_w1 = (const float*)d_in[15];
  const float* gl_b1 = (const float*)d_in[16];
  const float* gl_w2 = (const float*)d_in[17];
  const float* gl_b2 = (const float*)d_in[18];
  const float* gl_v1 = (const float*)d_in[19];
  const float* gl_c1 = (const float*)d_in[20];
  const float* gl_v2 = (const float*)d_in[21];
  const float* gl_c2 = (const float*)d_in[22];
  const float* le_w1 = (const float*)d_in[23];
  const float* le_b1 = (const float*)d_in[24];
  const float* le_w2 = (const float*)d_in[25];
  const float* le_b2 = (const float*)d_in[26];
  const float* le_w3 = (const float*)d_in[27];
  const float* le_b3 = (const float*)d_in[28];

  char* base = (char*)d_ws;
  size_t off = 0;
  auto take = [&](size_t bytes) -> char* {
    char* p = base + off;
    off = (off + bytes + 255) & ~(size_t)255;
    return p;
  };

  u16t*  objfA  = (u16t*) take((size_t)O_N * OBJP * 2);
  u16t*  objfB  = (u16t*) take((size_t)O_N * OBJP * 2);
  u16t*  edgfA  = (u16t*) take((size_t)T_N * 128 * 2);
  u16t*  edgfB  = (u16t*) take((size_t)T_N * 128 * 2);
  float* pooled = (float*)take((size_t)O_N * H_N * 4);
  float* invc   = (float*)take((size_t)O_N * 4);
  u16t *W1T[5], *W2T[5], *V1T[5], *V2T[5];
  for (int l = 0; l < 5; ++l) {
    W1T[l] = (u16t*)take((size_t)512 * KX * 2);
    W2T[l] = (u16t*)take((size_t)M_N * H_N * 2);
    V1T[l] = (u16t*)take((size_t)H_N * H_N * 2);
    V2T[l] = (u16t*)take((size_t)DOUT * H_N * 2);
  }
  u16t* L1T = (u16t*)take((size_t)128 * 128 * 2);
  u16t* L2T = (u16t*)take((size_t)128 * 128 * 2);
  u16t* L3T = (u16t*)take((size_t)16 * 128 * 2);

  auto blocks = [](long n) { return (int)((n + 255) / 256); };

  // --- weight prep (bf16 convert + transpose + K-pad) ---
  for (int l = 0; l < 5; ++l) {
    const float* w1 = (l == 0) ? g0_w1 : gl_w1 + (size_t)(l - 1) * 384 * 512;
    const float* w2 = (l == 0) ? g0_w2 : gl_w2 + (size_t)(l - 1) * 512 * 1152;
    const float* v1 = (l == 0) ? g0_v1 : gl_v1 + (size_t)(l - 1) * 512 * 512;
    const float* v2 = (l == 0) ? g0_v2 : gl_v2 + (size_t)(l - 1) * 512 * 128;
    k_build_w1t<<<blocks(512L * KX), 256, 0, stream>>>(w1, W1T[l], (l == 0) ? 143 : 128);
    k_transpose<<<blocks(512L * 1152), 256, 0, stream>>>(w2, W2T[l], 512, 1152);
    k_transpose<<<blocks(512L * 512),  256, 0, stream>>>(v1, V1T[l], 512, 512);
    k_transpose<<<blocks(512L * 128),  256, 0, stream>>>(v2, V2T[l], 512, 128);
  }
  k_transpose<<<blocks(128L * 128), 256, 0, stream>>>(le_w1, L1T, 128, 128);
  k_transpose<<<blocks(128L * 128), 256, 0, stream>>>(le_w2, L2T, 128, 128);
  k_transpose<<<blocks(128L * 16),  256, 0, stream>>>(le_w3, L3T, 128, 16);

  // --- graph-constant prep ---
  k_zero_f32<<<blocks(O_N), 256, 0, stream>>>(invc, O_N);
  k_counts<<<blocks(T_N), 256, 0, stream>>>(edges, invc);
  k_invert<<<blocks(O_N), 256, 0, stream>>>(invc);
  k_nodefeat0<<<blocks((long)O_N * OBJP), 256, 0, stream>>>(objs, pc, objfA);
  k_edgefeat0<<<blocks((long)T_N * 128), 256, 0, stream>>>(state, direct, emb_state, emb_drt, edgfA);

  // --- 5 gconv layers ---
  u16t *oin = objfA, *oout = objfB, *ein = edgfA, *eout = edgfB;
  for (int l = 0; l < 5; ++l) {
    const float* b1 = (l == 0) ? g0_b1 : gl_b1 + (size_t)(l - 1) * 512;
    const float* b2 = (l == 0) ? g0_b2 : gl_b2 + (size_t)(l - 1) * 1152;
    const float* c1 = (l == 0) ? g0_c1 : gl_c1 + (size_t)(l - 1) * 512;
    const float* c2 = (l == 0) ? g0_c2 : gl_c2 + (size_t)(l - 1) * 128;
    k_zero_f32<<<blocks((long)O_N * H_N), 256, 0, stream>>>(pooled, O_N * H_N);
    k_edge_gconv<<<T_N / BT, 256, 0, stream>>>(oin, ein, edges, W1T[l], b1,
                                               W2T[l], b2, pooled, eout);
    k_node_mlp<<<(O_N + 15) / 16, 256, 0, stream>>>(pooled, invc, V1T[l], c1,
                                                    V2T[l], c2, oout);
    u16t* t;
    t = oin; oin = oout; oout = t;
    t = ein; ein = eout; eout = t;
  }

  // --- final per-edge head ---
  k_edge_head<<<T_N / 16, 256, 0, stream>>>(ein, L1T, le_b1, L2T, le_b2,
                                            L3T, le_b3, (float*)d_out);
}
